// Detector_85315230368373
// MI455X (gfx1250) — compile-verified, matrix operand order checked by reading, not run on hardware
//
#include <hip/hip_runtime.h>
#include <hip/hip_bf16.h>

// ---------------------------------------------------------------------------
// Shapes (fixed by the reference)
// ---------------------------------------------------------------------------
#define BB    8
#define LL    24
#define TT    400
#define DD    1024
#define HH    512
#define GG    4
#define NEXP  (BB*LL)      // 192
#define NTOK  (NEXP*TT)    // 76800
#define RHD   128
#define HDD   256          // head dim
#define NHEAD 2
#define TEMPR 1.75f
#define ALPHB 0.5f

typedef __attribute__((ext_vector_type(16))) __bf16 v16bf;
typedef __attribute__((ext_vector_type(8)))  float  v8f;
typedef __attribute__((ext_vector_type(4)))  int    v4i;

union FragBF { uint4 u[2]; v16bf v; };

__device__ __forceinline__ float gelu_f(float x) {
    return 0.5f * x * (1.0f + erff(x * 0.7071067811865475f));
}

// ---------------------------------------------------------------------------
// CDNA5 async global->LDS copy (ASYNCcnt) with safe fallback
// ---------------------------------------------------------------------------
#if __has_builtin(__builtin_amdgcn_global_load_async_to_lds_b128)
#define HAVE_ASYNC_LDS 1
#else
#define HAVE_ASYNC_LDS 0
#endif

#if __has_builtin(__builtin_amdgcn_s_wait_asynccnt)
#define WAIT_ASYNC(n) __builtin_amdgcn_s_wait_asynccnt(n)
#else
#define WAIT_ASYNC(n) asm volatile("s_wait_asynccnt %0" :: "i"(n) : "memory")
#endif

__device__ __forceinline__ void async_copy16(const __hip_bfloat16* g, __hip_bfloat16* l) {
#if HAVE_ASYNC_LDS
    __builtin_amdgcn_global_load_async_to_lds_b128(
        (__attribute__((address_space(1))) v4i*)g,
        (__attribute__((address_space(3))) v4i*)l, 0, 0);
#else
    *(uint4*)l = *(const uint4*)g;
#endif
}

// lane layout for 16-bit WMMA A/B fragments (16x32 per CDNA5 ISA):
// lane m = lane&15 holds row m; hi = lane>>4 selects K halves:
// elems 0..7  -> K = k0 + hi*8 + j        (16 contiguous bytes)
// elems 8..15 -> K = k0 + 16 + hi*8 + j   (16 contiguous bytes)
__device__ __forceinline__ void load_frag(FragBF& f,
                                          const __hip_bfloat16* __restrict__ base,
                                          int ld, int row, int k0, int hi) {
    const char* p = (const char*)(base + (size_t)row * ld + k0 + hi * 8);
    f.u[0] = *(const uint4*)(p);
    f.u[1] = *(const uint4*)(p + 32);
}

__device__ __forceinline__ v8f wmma_bf16(const v16bf& A, const v16bf& B, v8f C) {
    return __builtin_amdgcn_wmma_f32_16x16x32_bf16(false, A, false, B, (short)0, C,
                                                   false, false);
}

// ---------------------------------------------------------------------------
// Weight prep: fold LN gain into transposed bf16 weights; fold LN bias into bias
// ---------------------------------------------------------------------------
__global__ void foldw_kernel(const float* __restrict__ W, const float* __restrict__ g,
                             __hip_bfloat16* __restrict__ WT, int Dk, int N) {
    int grp = blockIdx.y;
    const float* Wg = W + (size_t)grp * Dk * N;
    __hip_bfloat16* O = WT + (size_t)grp * Dk * N;
    int idx = blockIdx.x * 256 + threadIdx.x;
    if (idx >= Dk * N) return;
    int d = idx / N, n = idx % N;
    float v = Wg[idx];
    if (g) v *= g[(size_t)grp * Dk + d];
    O[(size_t)n * Dk + d] = __float2bfloat16(v);
}

__global__ void foldb_kernel(const float* __restrict__ W, const float* __restrict__ lnb,
                             const float* __restrict__ b, float* __restrict__ out,
                             int Dk, int N) {
    int grp = blockIdx.y;
    int n = blockIdx.x * 256 + threadIdx.x;
    if (n >= N) return;
    const float* Wg = W + (size_t)grp * Dk * N;
    const float* lb = lnb + (size_t)grp * Dk;
    float acc = b[(size_t)grp * N + n];
    for (int d = 0; d < Dk; ++d) acc += lb[d] * Wg[(size_t)d * N + n];
    out[(size_t)grp * N + n] = acc;
}

// LayerNorm (no affine) -> bf16 x_hat, one block per row
__global__ __launch_bounds__(256)
void ln_hat_kernel(const float* __restrict__ x, __hip_bfloat16* __restrict__ xhat,
                   int cols) {
    __shared__ float sh[256], sh2[256];
    size_t row = blockIdx.x;
    int tid = threadIdx.x;
    const float* xr = x + row * cols;
    float s = 0.f, s2 = 0.f;
    for (int c = tid; c < cols; c += 256) { float v = xr[c]; s += v; s2 += v * v; }
    sh[tid] = s; sh2[tid] = s2; __syncthreads();
    for (int off = 128; off; off >>= 1) {
        if (tid < off) { sh[tid] += sh[tid + off]; sh2[tid] += sh2[tid + off]; }
        __syncthreads();
    }
    float mu = sh[0] / cols;
    float var = sh2[0] / cols - mu * mu;
    float rs = rsqrtf(var + 1e-5f);
    __hip_bfloat16* o = xhat + row * cols;
    for (int c = tid; c < cols; c += 256) o[c] = __float2bfloat16((xr[c] - mu) * rs);
}

// ---------------------------------------------------------------------------
// Expert-batched WMMA GEMM: C[z][M,N] = epi( A[z][M,K] @ WT[g]^T + bias[g] )
// EPI 0: gelu -> bf16 | EPI 1: fp32 | EPI 2: resid + alpha*val -> fp32 | EPI 3: bf16
// Block tile 64(M) x 128(N); 8 waves as 4(M) x 2(N); wave = 16x64 (4 WMMA acc).
// Weight tile double-buffered in LDS via async global->LDS; A streamed direct.
// LDS B row stride 40 bf16 (80 B) -> conflict-free ds_load_b128.
// ---------------------------------------------------------------------------
template<int EPI>
__global__ __launch_bounds__(256)
void gemm_bf16_kernel(const __hip_bfloat16* __restrict__ A,
                      const __hip_bfloat16* __restrict__ WT,   // [G][N,K] transposed
                      const float* __restrict__ bias,          // [G][N]
                      __hip_bfloat16* __restrict__ outB,
                      float* __restrict__ outF,
                      const float* __restrict__ resid,
                      const float* __restrict__ alpha,
                      int M, int K, int N, int Lmod, int groupDiv) {
    __shared__ __hip_bfloat16 Bs[2][128 * 40];

    int z = blockIdx.z;
    int tid = threadIdx.x, lane = tid & 31, wave = tid >> 5;
    int wm = wave & 3, wn = wave >> 2;
    int m0 = blockIdx.y * 64 + wm * 16;
    int nblk = blockIdx.x * 128;
    bool mvalid = (m0 < M);

    int g = (groupDiv > 0) ? ((z % Lmod) / groupDiv) : 0;
    float al = (EPI == 2) ? alpha[z % Lmod] : 0.f;

    const __hip_bfloat16* Ae = A + (size_t)z * M * K;
    const __hip_bfloat16* Bt = WT + (size_t)g * K * N + (size_t)nblk * K;
    const float* bg = bias + (size_t)g * N;

    int mrow = lane & 15, hi = lane >> 4;
    int r = m0 + mrow; if (r >= M) r = M - 1;

    // thread's two 16B chunks of the 128x32 weight tile
    int c0 = tid, c1 = tid + 256;
    int cr0 = c0 >> 2, cq0 = c0 & 3;
    int cr1 = c1 >> 2, cq1 = c1 & 3;

    v8f accv[4] = {v8f{}, v8f{}, v8f{}, v8f{}};
    int nK = K >> 5;

    // stage 0 prefetch
    async_copy16(Bt + (size_t)cr0 * K + cq0 * 8, &Bs[0][cr0 * 40 + cq0 * 8]);
    async_copy16(Bt + (size_t)cr1 * K + cq1 * 8, &Bs[0][cr1 * 40 + cq1 * 8]);

    for (int kt = 0; kt < nK; ++kt) {
        int cur = kt & 1;
        int k0 = kt << 5;
        if (kt + 1 < nK) {
            int kn = (kt + 1) << 5;
            async_copy16(Bt + (size_t)cr0 * K + kn + cq0 * 8, &Bs[cur ^ 1][cr0 * 40 + cq0 * 8]);
            async_copy16(Bt + (size_t)cr1 * K + kn + cq1 * 8, &Bs[cur ^ 1][cr1 * 40 + cq1 * 8]);
            WAIT_ASYNC(2);
        } else {
            WAIT_ASYNC(0);
        }
        __syncthreads();
        if (mvalid) {
            FragBF fa;
            load_frag(fa, Ae, K, r, k0, hi);
            if (kt + 1 < nK)
                __builtin_prefetch((const void*)(Ae + (size_t)r * K + k0 + 32), 0, 1);
            const __hip_bfloat16* bsrc = &Bs[cur][0];
#pragma unroll
            for (int j = 0; j < 4; ++j) {
                FragBF fb;
                const char* p = (const char*)(bsrc + (wn * 64 + j * 16 + mrow) * 40 + hi * 8);
                fb.u[0] = *(const uint4*)p;
                fb.u[1] = *(const uint4*)(p + 32);
                accv[j] = wmma_bf16(fa.v, fb.v, accv[j]);
            }
        }
        __syncthreads();
    }

    if (!mvalid) return;
#pragma unroll
    for (int j = 0; j < 4; ++j) {
        int cn = nblk + wn * 64 + j * 16 + mrow;
        float bv = bg[cn];
#pragma unroll
        for (int e = 0; e < 8; ++e) {
            int ro = m0 + e + hi * 8;
            if (ro >= M) continue;
            size_t idx = ((size_t)z * M + ro) * N + cn;
            float v = accv[j][e] + bv;
            if (EPI == 0)      outB[idx] = __float2bfloat16(gelu_f(v));
            else if (EPI == 1) outF[idx] = v;
            else if (EPI == 2) outF[idx] = resid[idx] + al * v;
            else               outB[idx] = __float2bfloat16(v);
        }
    }
}

// ---------------------------------------------------------------------------
// Attention scores: per (n, head, t-tile of 16) compute full S rows with WMMA,
// stable softmax in LDS, accumulate column sums into wsum[n,head,s].
// ---------------------------------------------------------------------------
__global__ __launch_bounds__(256)
void attn_scores_kernel(const __hip_bfloat16* __restrict__ Q,
                        const __hip_bfloat16* __restrict__ Km,
                        float* __restrict__ wsum) {
    __shared__ float S[16][TT + 8];
    __shared__ float red[16][16];
    int z = blockIdx.z, head = blockIdx.y, tt = blockIdx.x;
    int tid = threadIdx.x, lane = tid & 31, wave = tid >> 5;
    int hi = lane >> 4, mrow = lane & 15;
    const __hip_bfloat16* Qb = Q + (size_t)z * TT * HH + head * HDD;
    const __hip_bfloat16* Kb = Km + (size_t)z * TT * HH + head * HDD;
    int t0 = tt * 16;
    const int nST = TT / 16;  // 25

    for (int st = wave; st < nST; st += 8) {
        v8f acc = {};
        for (int k0 = 0; k0 < HDD; k0 += 32) {
            FragBF fq, fk;
            load_frag(fq, Qb, HH, t0 + mrow, k0, hi);
            load_frag(fk, Kb, HH, st * 16 + mrow, k0, hi);
            acc = wmma_bf16(fq.v, fk.v, acc);
        }
        for (int e = 0; e < 8; ++e)
            S[e + hi * 8][st * 16 + mrow] = acc[e] * 0.0625f;  // 1/sqrt(256)
    }
    __syncthreads();

    int rrow = tid >> 4, rcol = tid & 15;
    float mx = -3.4e38f;
    for (int s = rcol; s < TT; s += 16) mx = fmaxf(mx, S[rrow][s]);
    red[rrow][rcol] = mx; __syncthreads();
    for (int off = 8; off; off >>= 1) {
        if (rcol < off) red[rrow][rcol] = fmaxf(red[rrow][rcol], red[rrow][rcol + off]);
        __syncthreads();
    }
    float rowmax = red[rrow][0]; __syncthreads();
    float sm = 0.f;
    for (int s = rcol; s < TT; s += 16) {
        float e = __expf(S[rrow][s] - rowmax); S[rrow][s] = e; sm += e;
    }
    red[rrow][rcol] = sm; __syncthreads();
    for (int off = 8; off; off >>= 1) {
        if (rcol < off) red[rrow][rcol] += red[rrow][rcol + off];
        __syncthreads();
    }
    float inv = 1.f / red[rrow][0]; __syncthreads();
    for (int s = rcol; s < TT; s += 16) S[rrow][s] *= inv;
    __syncthreads();

    float* wdst = wsum + ((size_t)z * NHEAD + head) * TT;
    for (int s = tid; s < TT; s += 256) {
        float cs = 0.f;
        for (int t = 0; t < 16; ++t) cs += S[t][s];
        atomicAdd(&wdst[s], cs);
    }
}

// Obar[n, head*HDD + d] = sum_s wsum[n,head,s] * V[n,s,head*HDD+d] / T
__global__ __launch_bounds__(256)
void attn_out_kernel(const __hip_bfloat16* __restrict__ V,
                     const float* __restrict__ wsum, float* __restrict__ obar) {
    int z = blockIdx.z, head = blockIdx.y, d = threadIdx.x;  // d in [0,256)
    const __hip_bfloat16* Vb = V + (size_t)z * TT * HH + head * HDD;
    const float* w = wsum + ((size_t)z * NHEAD + head) * TT;
    float acc = 0.f;
    for (int s = 0; s < TT; ++s)
        acc += w[s] * __bfloat162float(Vb[(size_t)s * HH + d]);
    obar[(size_t)z * HH + head * HDD + d] = acc * (1.0f / TT);
}

// small fp32 GEMM: out[m,n] = A[m,:]·W[:,n] + b[n]
__global__ void sgemm_f32_kernel(const float* __restrict__ A, const float* __restrict__ W,
                                 const float* __restrict__ b, float* __restrict__ out,
                                 int M, int K, int N) {
    int idx = blockIdx.x * 256 + threadIdx.x;
    if (idx >= M * N) return;
    int m = idx / N, n = idx % N;
    float acc = b ? b[n] : 0.f;
    const float* a = A + (size_t)m * K;
    for (int k = 0; k < K; ++k) acc += a[k] * W[(size_t)k * N + n];
    out[idx] = acc;
}

__global__ __launch_bounds__(128)
void router_kernel(const float* __restrict__ tp, const float* __restrict__ w1,
                   const float* __restrict__ b1, const float* __restrict__ w2,
                   const float* __restrict__ b2, float* __restrict__ rw_out) {
    int n = blockIdx.x, j = threadIdx.x;
    const float* x = tp + (size_t)n * HH;
    float acc = b1[j];
    for (int k = 0; k < HH; ++k) acc += x[k] * w1[(size_t)k * RHD + j];
    acc = gelu_f(acc);
    __shared__ float sh[128];
    sh[j] = acc * w2[j];
    __syncthreads();
    for (int off = 64; off; off >>= 1) {
        if (j < off) sh[j] += sh[j + off];
        __syncthreads();
    }
    if (j == 0) {
        float learned = 1.f / (1.f + __expf(-(sh[0] + b2[0]) / TEMPR));
        rw_out[n] = ALPHB * learned + (1.f - ALPHB) / LL;
    }
}

// fusion[b,t,:] = sum_l rw[b,l]*weighted[b,l,t,:]; LN; atomic mean over t
__global__ __launch_bounds__(256)
void fusion_kernel(const float* __restrict__ weighted, const float* __restrict__ rw,
                   const float* __restrict__ png, const float* __restrict__ pnb,
                   float* __restrict__ facc) {
    int t = blockIdx.x, b = blockIdx.y, tid = threadIdx.x;
    int h0 = tid, h1 = tid + 256;
    float f0 = 0.f, f1 = 0.f;
    for (int l = 0; l < LL; ++l) {
        float r = rw[b * LL + l];
        const float* wr = weighted + (((size_t)(b * LL + l)) * TT + t) * HH;
        f0 += r * wr[h0]; f1 += r * wr[h1];
    }
    __shared__ float sh[256], sh2[256];
    sh[tid] = f0 + f1; sh2[tid] = f0 * f0 + f1 * f1; __syncthreads();
    for (int off = 128; off; off >>= 1) {
        if (tid < off) { sh[tid] += sh[tid + off]; sh2[tid] += sh2[tid + off]; }
        __syncthreads();
    }
    float mu = sh[0] * (1.f / HH), var = sh2[0] * (1.f / HH) - mu * mu;
    float rs = rsqrtf(var + 1e-5f);
    float y0 = (f0 - mu) * rs * png[h0] + pnb[h0];
    float y1 = (f1 - mu) * rs * png[h1] + pnb[h1];
    atomicAdd(&facc[(size_t)b * HH + h0], y0 * (1.f / TT));
    atomicAdd(&facc[(size_t)b * HH + h1], y1 * (1.f / TT));
}

__global__ void meanl_kernel(const float* __restrict__ sh, float* __restrict__ sm) {
    size_t idx = (size_t)blockIdx.x * 256 + threadIdx.x;
    size_t tot = (size_t)BB * TT * HH;
    if (idx >= tot) return;
    int h = idx % HH; size_t bt = idx / HH; int t = bt % TT; int b = bt / TT;
    float acc = 0.f;
    for (int l = 0; l < LL; ++l)
        acc += sh[(((size_t)(b * LL + l)) * TT + t) * HH + h];
    sm[idx] = acc * (1.f / LL);
}

__global__ void zero_kernel(float* __restrict__ p, int n) {
    int i = blockIdx.x * 256 + threadIdx.x;
    if (i < n) p[i] = 0.f;
}

__global__ __launch_bounds__(256)
void final_kernel(const float* __restrict__ facc, const float* __restrict__ ctxv,
                  const float* __restrict__ pfg, const float* __restrict__ pfb,
                  const float* __restrict__ cw1, const float* __restrict__ cb1,
                  const float* __restrict__ cw2, const float* __restrict__ cb2,
                  const float* __restrict__ cw3, const float* __restrict__ cb3,
                  float* __restrict__ fused_out, float* __restrict__ logits_out) {
    int b = blockIdx.x, tid = threadIdx.x;
    __shared__ float sh[256], sh2[256];
    __shared__ float fs[512], c1s[256], c2s[128];

    float a0 = facc[b * 512 + tid], a1 = facc[b * 512 + 256 + tid];
    sh[tid] = a0 + a1; sh2[tid] = a0 * a0 + a1 * a1; __syncthreads();
    for (int off = 128; off; off >>= 1) {
        if (tid < off) { sh[tid] += sh[tid + off]; sh2[tid] += sh2[tid + off]; }
        __syncthreads();
    }
    float mu = sh[0] * (1.f / 512.f), var = sh2[0] * (1.f / 512.f) - mu * mu;
    float rs = rsqrtf(var + 1e-5f);
    float l0 = (a0 - mu) * rs * pfg[tid] + pfb[tid];
    float l1 = (a1 - mu) * rs * pfg[256 + tid] + pfb[256 + tid];
    __syncthreads();

    float g0 = ctxv[b * 512 + tid], g1 = ctxv[b * 512 + 256 + tid];
    sh[tid] = g0 + g1; sh2[tid] = g0 * g0 + g1 * g1; __syncthreads();
    for (int off = 128; off; off >>= 1) {
        if (tid < off) { sh[tid] += sh[tid + off]; sh2[tid] += sh2[tid + off]; }
        __syncthreads();
    }
    float mu2 = sh[0] * (1.f / 512.f), var2 = sh2[0] * (1.f / 512.f) - mu2 * mu2;
    float rs2 = rsqrtf(var2 + 1e-5f);
    float m0 = (g0 - mu2) * rs2 * pfg[tid] + pfb[tid];
    float m1 = (g1 - mu2) * rs2 * pfg[256 + tid] + pfb[256 + tid];
    float f0 = l0 + m0, f1 = l1 + m1;
    fs[tid] = f0; fs[256 + tid] = f1;
    fused_out[b * 512 + tid] = f0; fused_out[b * 512 + 256 + tid] = f1;
    __syncthreads();

    float acc = cb1[tid];
    for (int k = 0; k < 512; ++k) acc += fs[k] * cw1[k * 256 + tid];
    c1s[tid] = gelu_f(acc); __syncthreads();
    if (tid < 128) {
        float acc2 = cb2[tid];
        for (int k = 0; k < 256; ++k) acc2 += c1s[k] * cw2[k * 128 + tid];
        c2s[tid] = gelu_f(acc2);
    }
    __syncthreads();
    if (tid < 2) {
        float acc3 = cb3[tid];
        for (int k = 0; k < 128; ++k) acc3 += c2s[k] * cw3[k * 2 + tid];
        logits_out[b * 2 + tid] = acc3;
    }
}

// ---------------------------------------------------------------------------
// Host orchestration
// ---------------------------------------------------------------------------
extern "C" void kernel_launch(void* const* d_in, const int* in_sizes, int n_in,
                              void* d_out, int out_size, void* d_ws, size_t ws_size,
                              hipStream_t stream) {
    (void)in_sizes; (void)n_in; (void)out_size; (void)ws_size;
    const float* x      = (const float*)d_in[0];
    const float* s_ln_g = (const float*)d_in[1];
    const float* s_ln_b = (const float*)d_in[2];
    const float* s_w1   = (const float*)d_in[3];
    const float* s_b1   = (const float*)d_in[4];
    const float* s_w2   = (const float*)d_in[5];
    const float* s_b2   = (const float*)d_in[6];
    const float* g_ln_g = (const float*)d_in[7];
    const float* g_ln_b = (const float*)d_in[8];
    const float* g_w1   = (const float*)d_in[9];
    const float* g_b1   = (const float*)d_in[10];
    const float* g_w2   = (const float*)d_in[11];
    const float* g_b2   = (const float*)d_in[12];
    const float* mlp_a  = (const float*)d_in[13];
    const float* p_ln_g = (const float*)d_in[14];
    const float* p_ln_b = (const float*)d_in[15];
    const float* p_wq   = (const float*)d_in[16];
    const float* p_wk   = (const float*)d_in[17];
    const float* p_wv   = (const float*)d_in[18];
    const float* p_bq   = (const float*)d_in[19];
    const float* p_bk   = (const float*)d_in[20];
    const float* p_bv   = (const float*)d_in[21];
    const float* p_wo   = (const float*)d_in[22];
    const float* p_bo   = (const float*)d_in[23];
    const float* r_w1   = (const float*)d_in[24];
    const float* r_b1   = (const float*)d_in[25];
    const float* r_w2   = (const float*)d_in[26];
    const float* r_b2   = (const float*)d_in[27];
    const float* pn_g   = (const float*)d_in[28];
    const float* pn_b   = (const float*)d_in[29];
    const float* pf_g   = (const float*)d_in[30];
    const float* pf_b   = (const float*)d_in[31];
    const float* c_w1   = (const float*)d_in[32];
    const float* c_b1   = (const float*)d_in[33];
    const float* c_w2   = (const float*)d_in[34];
    const float* c_b2   = (const float*)d_in[35];
    const float* c_w3   = (const float*)d_in[36];
    const float* c_b3   = (const float*)d_in[37];

    float* out        = (float*)d_out;
    float* out_logits = out;            // [8,2]
    float* out_rw     = out + 16;       // [8,24]
    float* out_fused  = out + 208;      // [8,512]
    float* out_tp     = out + 4304;     // [8,24,512]

    char* ws = (char*)d_ws;
    const size_t SZ_XHAT = (size_t)NTOK * DD * 2;   // 157,286,400
    const size_t SZ_H2   = (size_t)NTOK * HH * 2;   // 78,643,200
    const size_t SZ_H4   = (size_t)NTOK * HH * 4;   // 157,286,400
    size_t A_ = 0;
    size_t B_ = A_ + SZ_XHAT;
    size_t C_ = B_ + SZ_H2;
    size_t D_ = C_ + SZ_H4;
    size_t E_ = D_ + SZ_H4;
    size_t S_ = E_ + SZ_H2;

    __hip_bfloat16* xhat  = (__hip_bfloat16*)(ws + A_);
    __hip_bfloat16* qb    = (__hip_bfloat16*)(ws + A_);             // reuse
    __hip_bfloat16* kb    = (__hip_bfloat16*)(ws + A_ + SZ_H2);     // reuse
    __hip_bfloat16* y1    = (__hip_bfloat16*)(ws + B_);
    __hip_bfloat16* vb    = (__hip_bfloat16*)(ws + B_);             // reuse
    float*          sharedp   = (float*)(ws + C_);
    float*          weightedp = (float*)(ws + D_);
    __hip_bfloat16* whatp = (__hip_bfloat16*)(ws + E_);
    float*          smp   = (float*)(ws + E_);                      // reuse
    __hip_bfloat16* smhat = (__hip_bfloat16*)(ws + E_ + 6553600);
    __hip_bfloat16* q2    = (__hip_bfloat16*)(ws + E_ + 9830400);
    __hip_bfloat16* k2    = (__hip_bfloat16*)(ws + E_ + 13107200);
    __hip_bfloat16* v2    = (__hip_bfloat16*)(ws + E_ + 16384000);

    size_t o = S_;
    auto nxt = [&](size_t bytes) { size_t r = o; o = (o + bytes + 255) & ~(size_t)255; return r; };
    float* wsum_main = (float*)(ws + nxt(614400));   // [192,2,400]
    float* wsum_ctx  = (float*)(ws + nxt(25600));    // [8,2,400]
    float* facc      = (float*)(ws + nxt(16384));    // [8,512]  (zeroed together)
    float* obar      = (float*)(ws + nxt(393216));   // [192,512]
    float* obar2     = (float*)(ws + nxt(16384));    // [8,512]
    float* ctxv      = (float*)(ws + nxt(16384));    // [8,512]
    __hip_bfloat16* ws1T = (__hip_bfloat16*)(ws + nxt((size_t)DD * HH * 2));
    float*          bs1  = (float*)(ws + nxt(HH * 4));
    __hip_bfloat16* wg1T = (__hip_bfloat16*)(ws + nxt((size_t)GG * DD * HH * 2));
    float*          bg1  = (float*)(ws + nxt(GG * HH * 4));
    __hip_bfloat16* ws2T = (__hip_bfloat16*)(ws + nxt((size_t)HH * HH * 2));
    __hip_bfloat16* wg2T = (__hip_bfloat16*)(ws + nxt((size_t)GG * HH * HH * 2));
    __hip_bfloat16* wqT  = (__hip_bfloat16*)(ws + nxt((size_t)HH * HH * 2));
    __hip_bfloat16* wkT  = (__hip_bfloat16*)(ws + nxt((size_t)HH * HH * 2));
    __hip_bfloat16* wvT  = (__hip_bfloat16*)(ws + nxt((size_t)HH * HH * 2));
    float* bqF = (float*)(ws + nxt(HH * 4));
    float* bkF = (float*)(ws + nxt(HH * 4));
    float* bvF = (float*)(ws + nxt(HH * 4));

    dim3 blk(256);

    // ---- weight prep (tiny) ----
    foldw_kernel<<<dim3((DD * HH + 255) / 256, 1), blk, 0, stream>>>(s_w1, s_ln_g, ws1T, DD, HH);
    foldb_kernel<<<dim3((HH + 255) / 256, 1), blk, 0, stream>>>(s_w1, s_ln_b, s_b1, bs1, DD, HH);
    foldw_kernel<<<dim3((DD * HH + 255) / 256, GG), blk, 0, stream>>>(g_w1, g_ln_g, wg1T, DD, HH);
    foldb_kernel<<<dim3((HH + 255) / 256, GG), blk, 0, stream>>>(g_w1, g_ln_b, g_b1, bg1, DD, HH);
    foldw_kernel<<<dim3((HH * HH + 255) / 256, 1), blk, 0, stream>>>(s_w2, nullptr, ws2T, HH, HH);
    foldw_kernel<<<dim3((HH * HH + 255) / 256, GG), blk, 0, stream>>>(g_w2, nullptr, wg2T, HH, HH);
    foldw_kernel<<<dim3((HH * HH + 255) / 256, 1), blk, 0, stream>>>(p_wq, p_ln_g, wqT, HH, HH);
    foldw_kernel<<<dim3((HH * HH + 255) / 256, 1), blk, 0, stream>>>(p_wk, p_ln_g, wkT, HH, HH);
    foldw_kernel<<<dim3((HH * HH + 255) / 256, 1), blk, 0, stream>>>(p_wv, p_ln_g, wvT, HH, HH);
    foldb_kernel<<<dim3((HH + 255) / 256, 1), blk, 0, stream>>>(p_wq, p_ln_b, p_bq, bqF, HH, HH);
    foldb_kernel<<<dim3((HH + 255) / 256, 1), blk, 0, stream>>>(p_wk, p_ln_b, p_bk, bkF, HH, HH);
    foldb_kernel<<<dim3((HH + 255) / 256, 1), blk, 0, stream>>>(p_wv, p_ln_b, p_bv, bvF, HH, HH);

    // ---- main pipeline ----
    ln_hat_kernel<<<NTOK, blk, 0, stream>>>(x, xhat, DD);

    dim3 gG(HH / 128, (TT + 63) / 64, NEXP);  // (4,7,192)
    gemm_bf16_kernel<0><<<gG, blk, 0, stream>>>(xhat, ws1T, bs1, y1, nullptr, nullptr, nullptr, TT, DD, HH, LL, 0);
    gemm_bf16_kernel<1><<<gG, blk, 0, stream>>>(y1, ws2T, s_b2, nullptr, sharedp, nullptr, nullptr, TT, HH, HH, LL, 0);
    gemm_bf16_kernel<0><<<gG, blk, 0, stream>>>(xhat, wg1T, bg1, y1, nullptr, nullptr, nullptr, TT, DD, HH, LL, LL / GG);
    gemm_bf16_kernel<2><<<gG, blk, 0, stream>>>(y1, wg2T, g_b2, nullptr, weightedp, sharedp, mlp_a, TT, HH, HH, LL, LL / GG);

    ln_hat_kernel<<<NTOK, blk, 0, stream>>>(weightedp, whatp, HH);

    gemm_bf16_kernel<3><<<gG, blk, 0, stream>>>(whatp, wqT, bqF, qb, nullptr, nullptr, nullptr, TT, HH, HH, LL, 0);
    gemm_bf16_kernel<3><<<gG, blk, 0, stream>>>(whatp, wkT, bkF, kb, nullptr, nullptr, nullptr, TT, HH, HH, LL, 0);
    gemm_bf16_kernel<3><<<gG, blk, 0, stream>>>(whatp, wvT, bvF, vb, nullptr, nullptr, nullptr, TT, HH, HH, LL, 0);

    zero_kernel<<<(164096 + 255) / 256, blk, 0, stream>>>(wsum_main, 164096);

    attn_scores_kernel<<<dim3(TT / 16, NHEAD, NEXP), blk, 0, stream>>>(qb, kb, wsum_main);
    attn_out_kernel<<<dim3(1, NHEAD, NEXP), blk, 0, stream>>>(vb, wsum_main, obar);

    sgemm_f32_kernel<<<(NEXP * HH + 255) / 256, blk, 0, stream>>>(obar, p_wo, p_bo, out_tp, NEXP, HH, HH);
    router_kernel<<<NEXP, 128, 0, stream>>>(out_tp, r_w1, r_b1, r_w2, r_b2, out_rw);
    fusion_kernel<<<dim3(TT, BB), blk, 0, stream>>>(weightedp, out_rw, pn_g, pn_b, facc);

    // ---- ctx path: mean over L of shared, then attention pool on [8,400,512] ----
    meanl_kernel<<<((size_t)BB * TT * HH + 255) / 256, blk, 0, stream>>>(sharedp, smp);
    ln_hat_kernel<<<BB * TT, blk, 0, stream>>>(smp, smhat, HH);
    dim3 gC(HH / 128, (TT + 63) / 64, BB);
    gemm_bf16_kernel<3><<<gC, blk, 0, stream>>>(smhat, wqT, bqF, q2, nullptr, nullptr, nullptr, TT, HH, HH, LL, 0);
    gemm_bf16_kernel<3><<<gC, blk, 0, stream>>>(smhat, wkT, bkF, k2, nullptr, nullptr, nullptr, TT, HH, HH, LL, 0);
    gemm_bf16_kernel<3><<<gC, blk, 0, stream>>>(smhat, wvT, bvF, v2, nullptr, nullptr, nullptr, TT, HH, HH, LL, 0);
    attn_scores_kernel<<<dim3(TT / 16, NHEAD, BB), blk, 0, stream>>>(q2, k2, wsum_ctx);
    attn_out_kernel<<<dim3(1, NHEAD, BB), blk, 0, stream>>>(v2, wsum_ctx, obar2);
    sgemm_f32_kernel<<<(BB * HH + 255) / 256, blk, 0, stream>>>(obar2, p_wo, p_bo, ctxv, BB, HH, HH);

    final_kernel<<<BB, blk, 0, stream>>>(facc, ctxv, pf_g, pf_b, c_w1, c_b1, c_w2, c_b2,
                                         c_w3, c_b3, out_fused, out_logits);
}
